// GConv_49538152792945
// MI455X (gfx1250) — matrix operand order repaired and checked by reading, not compiled, not run on hardware
//
#include <hip/hip_runtime.h>

// ---------------------------------------------------------------------------
// SE(3)-transformer-style graph conv for MI455X (gfx1250, wave32, WMMA).
// All MLP GEMMs run through v_wmma_f32_16x16x32_f16 (f16 in, f32 accum).
// BN stats are produced in the GEMM epilogue (LDS reduce + atomics);
// BN+ReLU+f16-convert is fused into the A-fragment load of the next GEMM.
// ---------------------------------------------------------------------------

typedef __attribute__((ext_vector_type(16))) _Float16 v16h;
typedef __attribute__((ext_vector_type(8)))  _Float16 v8h;
typedef __attribute__((ext_vector_type(8)))  float    v8f;

union F16x16 { v16h v; v8h h[2]; };

constexpr int   kN  = 16000;
constexpr int   kE  = 160000;
constexpr float kEPS = 1e-5f;

static inline int cdiv(long a, int b) { return (int)((a + b - 1) / b); }

// ---------------------------------------------------------------------------
// Shared epilogue: store C tile (+bias) and optionally accumulate per-column
// sum / sumsq for BatchNorm into stats[2*NOUT] via LDS reduction.
// Requires M % 128 == 0 (all waves in all blocks active).
// ---------------------------------------------------------------------------
template<int NOUT, bool BIAS, bool STATS, int NT>
__device__ __forceinline__
void gemm_epilogue(v8f (&acc)[NT], const float* __restrict__ bias,
                   float* __restrict__ C, float* __restrict__ stats,
                   int m0, int nlo, int hi)
{
    #pragma unroll
    for (int t = 0; t < NT; ++t) {
        const int col = t * 16 + nlo;
        const float bv = BIAS ? bias[col] : 0.f;
        #pragma unroll
        for (int r = 0; r < 8; ++r) {
            const int row = m0 + r + 8 * hi;   // C layout: VGPR r -> M = r + 8*hi
            C[(size_t)row * NOUT + col] = acc[t][r] + bv;
        }
    }
    if constexpr (STATS) {
        __shared__ float sb[2 * NOUT];
        for (int i = threadIdx.x; i < 2 * NOUT; i += 256) sb[i] = 0.f;
        __syncthreads();
        #pragma unroll
        for (int t = 0; t < NT; ++t) {
            const int col = t * 16 + nlo;
            float ps = 0.f, pq = 0.f;
            #pragma unroll
            for (int r = 0; r < 8; ++r) { float x = acc[t][r]; ps += x; pq += x * x; }
            atomicAdd(&sb[col], ps);
            atomicAdd(&sb[NOUT + col], pq);
        }
        __syncthreads();
        for (int i = threadIdx.x; i < NOUT; i += 256) {
            atomicAdd(&stats[i],        sb[i]);
            atomicAdd(&stats[NOUT + i], sb[NOUT + i]);
        }
    }
}

// ---------------------------------------------------------------------------
// WMMA GEMM, f16 A: C[M,NOUT] = X[M,KP] @ W[NOUT,KP]^T (+bias) [+stats]
// KP mult of 32, NOUT mult of 16, M mult of 128. 256 thr = 8 waves/block.
// ---------------------------------------------------------------------------
template<int KP, int NOUT, bool BIAS, bool STATS>
__global__ void __launch_bounds__(256)
wmma_gemm_xwT(const _Float16* __restrict__ X, const _Float16* __restrict__ W,
              const float* __restrict__ bias, float* __restrict__ C,
              float* __restrict__ stats, int M)
{
    constexpr int NT = NOUT / 16;
    const int wave = threadIdx.x >> 5;
    const int lane = threadIdx.x & 31;
    const int m0   = (blockIdx.x * 8 + wave) * 16;
    const int nlo  = lane & 15;
    const int hi   = lane >> 4;
    const int mrow = m0 + nlo;

    v8f acc[NT] = {};
    #pragma unroll
    for (int kb = 0; kb < KP; kb += 32) {
        // A 16x32 f16: halves 0..7 = K in [kb+hi*8,+8); 8..15 = [kb+16+hi*8,+8)
        F16x16 a;
        a.h[0] = *(const v8h*)(X + (size_t)mrow * KP + kb + hi * 8);
        a.h[1] = *(const v8h*)(X + (size_t)mrow * KP + kb + 16 + hi * 8);
        #pragma unroll
        for (int t = 0; t < NT; ++t) {
            const int n = t * 16 + nlo;   // B 32x16: halves 0..7 = K in [kb+hi*8,+8); 8..15 = [kb+16+hi*8,+8)
            F16x16 bu;
            bu.h[0] = *(const v8h*)(W + (size_t)n * KP + kb + hi * 8);
            bu.h[1] = *(const v8h*)(W + (size_t)n * KP + kb + 16 + hi * 8);
            v16h b = bu.v;
            acc[t] = __builtin_amdgcn_wmma_f32_16x16x32_f16(
                false, a.v, false, b, (short)0, acc[t], false, false);
        }
    }
    gemm_epilogue<NOUT, BIAS, STATS, NT>(acc, bias, C, stats, m0, nlo, hi);
}

// ---------------------------------------------------------------------------
// WMMA GEMM, fused-BN A: A = f16(relu(scale[k]*Xpre[m,k] + shift[k])).
// Xpre is the f32 pre-activation of the previous layer; avoids a separate
// activation buffer + full extra read/write pass.
// ---------------------------------------------------------------------------
template<int KP, int NOUT, bool BIAS, bool STATS>
__global__ void __launch_bounds__(256)
wmma_gemm_bnA(const float* __restrict__ Xpre, const float* __restrict__ scale,
              const float* __restrict__ shift, const _Float16* __restrict__ W,
              const float* __restrict__ bias, float* __restrict__ C,
              float* __restrict__ stats, int M)
{
    constexpr int NT = NOUT / 16;
    const int wave = threadIdx.x >> 5;
    const int lane = threadIdx.x & 31;
    const int m0   = (blockIdx.x * 8 + wave) * 16;
    const int nlo  = lane & 15;
    const int hi   = lane >> 4;
    const int mrow = m0 + nlo;

    v8f acc[NT] = {};
    #pragma unroll
    for (int kb = 0; kb < KP; kb += 32) {
        const int k0 = kb + hi * 8;        // halves 0..7
        const int k1 = kb + 16 + hi * 8;   // halves 8..15
        v8f x0 = *(const v8f*)(Xpre + (size_t)mrow * KP + k0);
        v8f x1 = *(const v8f*)(Xpre + (size_t)mrow * KP + k1);
        v8f s0 = *(const v8f*)(scale + k0);
        v8f s1 = *(const v8f*)(scale + k1);
        v8f h0 = *(const v8f*)(shift + k0);
        v8f h1 = *(const v8f*)(shift + k1);
        F16x16 a;
        #pragma unroll
        for (int j = 0; j < 8; ++j) {
            a.h[0][j] = (_Float16)fmaxf(s0[j] * x0[j] + h0[j], 0.f);
            a.h[1][j] = (_Float16)fmaxf(s1[j] * x1[j] + h1[j], 0.f);
        }
        #pragma unroll
        for (int t = 0; t < NT; ++t) {
            const int n = t * 16 + nlo;
            F16x16 bu;
            bu.h[0] = *(const v8h*)(W + (size_t)n * KP + k0);
            bu.h[1] = *(const v8h*)(W + (size_t)n * KP + k1);
            v16h b = bu.v;
            acc[t] = __builtin_amdgcn_wmma_f32_16x16x32_f16(
                false, a.v, false, b, (short)0, acc[t], false, false);
        }
    }
    gemm_epilogue<NOUT, BIAS, STATS, NT>(acc, bias, C, stats, m0, nlo, hi);
}

// ---------------------------------------------------------------------------
// f32 -> f16 weight conversion with zero padding. dst is [*, KP] row-major.
// ---------------------------------------------------------------------------
__global__ void cvt_pad_f16(const float* __restrict__ src, int NR, int KR,
                            _Float16* __restrict__ dst, int KP, int total)
{
    int i = blockIdx.x * blockDim.x + threadIdx.x;
    if (i >= total) return;
    int n = i / KP, k = i % KP;
    dst[i] = (n < NR && k < KR) ? (_Float16)src[n * KR + k] : (_Float16)0.f;
}

// ---------------------------------------------------------------------------
// mean/sumsq of dist over E (scalar batch-norm stats for all edge-MLP layer 1).
// ---------------------------------------------------------------------------
__global__ void __launch_bounds__(256)
reduce_dist_stats(const float* __restrict__ dist, float* __restrict__ out)
{
    __shared__ float ss[256], sq[256];
    float s = 0.f, q = 0.f;
    for (int i = blockIdx.x * 256 + threadIdx.x; i < kE; i += 256 * 256) {
        float d = dist[i]; s += d; q += d * d;
    }
    ss[threadIdx.x] = s; sq[threadIdx.x] = q;
    __syncthreads();
    for (int g = 128; g > 0; g >>= 1) {
        if ((int)threadIdx.x < g) {
            ss[threadIdx.x] += ss[threadIdx.x + g];
            sq[threadIdx.x] += sq[threadIdx.x + g];
        }
        __syncthreads();
    }
    if (threadIdx.x == 0) { atomicAdd(&out[0], ss[0]); atomicAdd(&out[1], sq[0]); }
}

// ---------------------------------------------------------------------------
// Edge MLP layer 1 (input = scalar dist): b1 cancels in BN; BN stats reduce to
// mean/var of dist. Cols 10..31 zero (K padding for the K=32 WMMA step).
// ---------------------------------------------------------------------------
__global__ void edge_layer1(const float* __restrict__ dist, const float* __restrict__ W1,
                            const float* __restrict__ g1, const float* __restrict__ be1,
                            const float* __restrict__ dstats, _Float16* __restrict__ h1)
{
    int e = blockIdx.x * blockDim.x + threadIdx.x;
    if (e >= kE) return;
    const float inv = 1.f / (float)kE;
    float mean = dstats[0] * inv;
    float var  = dstats[1] * inv - mean * mean;
    float dc = dist[e] - mean;
    _Float16* row = h1 + (size_t)e * 32;
    #pragma unroll
    for (int j = 0; j < 10; ++j) {
        float w = W1[j];
        float y = g1[j] * (w * dc) * rsqrtf(w * w * var + kEPS) + be1[j];
        row[j] = (_Float16)fmaxf(y, 0.f);
    }
    #pragma unroll
    for (int j = 10; j < 32; ++j) row[j] = (_Float16)0.f;
}

// Fold BN stats into per-column scale/shift; padded cols -> 0.
__global__ void bn_finalize(const float* __restrict__ stats, const float* __restrict__ g,
                            const float* __restrict__ be, int C, int CR, float invM,
                            float* __restrict__ scale, float* __restrict__ shift)
{
    int j = threadIdx.x;
    if (j >= C) return;
    if (j < CR) {
        float mean = stats[j] * invM;
        float var  = stats[C + j] * invM - mean * mean;
        float sc = g[j] * rsqrtf(var + kEPS);
        scale[j] = sc; shift[j] = be[j] - sc * mean;
    } else { scale[j] = 0.f; shift[j] = 0.f; }
}

// ---------------------------------------------------------------------------
// Per-edge CG contraction: tmp[j,i,ml] = sum_mk wj[e,j,ml,mk] * f[idx[e],i,mk]
//                          out[e,o,ml] += sum_{j,i} R[e,j,o,i] * tmp[j,i,ml]
// ---------------------------------------------------------------------------
template<int L, int K>
__global__ void edge_wlayer(const float* __restrict__ R, const float* __restrict__ wj,
                            const float* __restrict__ f, const int* __restrict__ idx,
                            float* __restrict__ out, int ldo, int col0)
{
    constexpr int ML = 2 * L + 1, MK = 2 * K + 1;
    constexpr int J  = 2 * ((L < K) ? L : K) + 1;
    int e = blockIdx.x * blockDim.x + threadIdx.x;
    if (e >= kE) return;
    int node = idx[e];
    float fe[8][MK];
    const float* fp = f + (size_t)node * 8 * MK;
    #pragma unroll
    for (int i = 0; i < 8; ++i)
        #pragma unroll
        for (int mk = 0; mk < MK; ++mk) fe[i][mk] = fp[i * MK + mk];

    const float* wp = wj + (size_t)e * (J * ML * MK);
    float tmp[J][8][ML];
    #pragma unroll
    for (int j = 0; j < J; ++j)
        #pragma unroll
        for (int i = 0; i < 8; ++i)
            #pragma unroll
            for (int ml = 0; ml < ML; ++ml) {
                float s = 0.f;
                #pragma unroll
                for (int mk = 0; mk < MK; ++mk) s += wp[(j * ML + ml) * MK + mk] * fe[i][mk];
                tmp[j][i][ml] = s;
            }
    const float* rp = R + (size_t)e * (J * 64);
    #pragma unroll
    for (int o = 0; o < 8; ++o) {
        float acc[ML];
        #pragma unroll
        for (int ml = 0; ml < ML; ++ml) acc[ml] = 0.f;
        #pragma unroll
        for (int j = 0; j < J; ++j)
            #pragma unroll
            for (int i = 0; i < 8; ++i) {
                float r = rp[(j * 8 + o) * 8 + i];
                #pragma unroll
                for (int ml = 0; ml < ML; ++ml) acc[ml] += r * tmp[j][i][ml];
            }
        float* op = out + ((size_t)e * 8 + o) * ldo + col0;
        #pragma unroll
        for (int ml = 0; ml < ML; ++ml) op[ml] += acc[ml];
    }
}

// q[n,o,0] = wq0 . f0 ; q[n,o,1..3] = wq1 . f1   (layout [N,8,4])
__global__ void node_q(const float* __restrict__ f0, const float* __restrict__ f1,
                       const float* __restrict__ wq, float* __restrict__ q)
{
    int n = blockIdx.x * blockDim.x + threadIdx.x;
    if (n >= kN) return;
    #pragma unroll
    for (int o = 0; o < 8; ++o) {
        float s0 = 0.f;
        #pragma unroll
        for (int i = 0; i < 8; ++i) s0 += wq[o * 8 + i] * f0[(size_t)n * 8 + i];
        q[((size_t)n * 8 + o) * 4 + 0] = s0;
        #pragma unroll
        for (int ml = 0; ml < 3; ++ml) {
            float s = 0.f;
            #pragma unroll
            for (int i = 0; i < 8; ++i) s += wq[64 + o * 8 + i] * f1[((size_t)n * 8 + i) * 3 + ml];
            q[((size_t)n * 8 + o) * 4 + 1 + ml] = s;
        }
    }
}

// ex[e] = exp(q[dst[e]] . k_e[e]); denom[dst] += ex
__global__ void edge_attn_ex(const float* __restrict__ q, const float* __restrict__ ke,
                             const int* __restrict__ dst, float* __restrict__ ex,
                             float* __restrict__ denom)
{
    int e = blockIdx.x * blockDim.x + threadIdx.x;
    if (e >= kE) return;
    int d = dst[e];
    const float* qp = q + (size_t)d * 32;
    const float* kp = ke + (size_t)e * 32;
    float s = 0.f;
    #pragma unroll
    for (int i = 0; i < 32; ++i) s += qp[i] * kp[i];
    float x = __expf(s);
    ex[e] = x;
    atomicAdd(&denom[d], x);
}

// segment-sum of a * v into node outputs (both degrees)
__global__ void edge_agg(const float* __restrict__ ex, const float* __restrict__ denom,
                         const int* __restrict__ dst, const float* __restrict__ v0,
                         const float* __restrict__ v1, float* __restrict__ out0,
                         float* __restrict__ out1)
{
    int e = blockIdx.x * blockDim.x + threadIdx.x;
    if (e >= kE) return;
    int d = dst[e];
    float a = ex[e] / denom[d];
    #pragma unroll
    for (int o = 0; o < 8; ++o) {
        atomicAdd(&out0[(size_t)d * 8 + o], a * v0[(size_t)e * 8 + o]);
        #pragma unroll
        for (int ml = 0; ml < 3; ++ml)
            atomicAdd(&out1[((size_t)d * 8 + o) * 3 + ml],
                      a * v1[((size_t)e * 8 + o) * 3 + ml]);
    }
}

// inner[n, c*8+d] = sum_ml f[n,c,ml]*f[n,d,ml]  -> f16 (K=64 GEMM input)
template<int ML>
__global__ void node_inner(const float* __restrict__ f, _Float16* __restrict__ inner)
{
    int n = blockIdx.x * blockDim.x + threadIdx.x;
    if (n >= kN) return;
    float fl[8][ML];
    #pragma unroll
    for (int c = 0; c < 8; ++c)
        #pragma unroll
        for (int ml = 0; ml < ML; ++ml) fl[c][ml] = f[((size_t)n * 8 + c) * ML + ml];
    #pragma unroll
    for (int c = 0; c < 8; ++c)
        #pragma unroll
        for (int d2 = 0; d2 < 8; ++d2) {
            float s = 0.f;
            #pragma unroll
            for (int ml = 0; ml < ML; ++ml) s += fl[c][ml] * fl[d2][ml];
            inner[(size_t)n * 64 + c * 8 + d2] = (_Float16)s;
        }
}

// out[n,o,ml] += sum_i siw[n,o,i] * f[n,i,ml]
template<int ML>
__global__ void node_final(const float* __restrict__ siw, const float* __restrict__ f,
                           float* __restrict__ out)
{
    int n = blockIdx.x * blockDim.x + threadIdx.x;
    if (n >= kN) return;
    #pragma unroll
    for (int o = 0; o < 8; ++o)
        #pragma unroll
        for (int ml = 0; ml < ML; ++ml) {
            float s = 0.f;
            #pragma unroll
            for (int i = 0; i < 8; ++i)
                s += siw[(size_t)n * 64 + o * 8 + i] * f[((size_t)n * 8 + i) * ML + ml];
            out[((size_t)n * 8 + o) * ML + ml] += s;
        }
}

// ---------------------------------------------------------------------------
// Host driver
// ---------------------------------------------------------------------------
extern "C" void kernel_launch(void* const* d_in, const int* in_sizes, int n_in,
                              void* d_out, int out_size, void* d_ws, size_t ws_size,
                              hipStream_t stream)
{
    (void)in_sizes; (void)n_in; (void)ws_size;

    // d_in: top-level in setup_inputs() insertion order; nested pytrees
    // flattened list-in-order / dict-in-sorted-key-order (JAX convention).
    // Per-MLP sorted keys: W1,W2,W3,b1,b2,b3,be1,be2,g1,g2.
    const float* f0   = (const float*)d_in[0];
    const float* f1   = (const float*)d_in[1];
    const float* dist = (const float*)d_in[2];
    const int*   src  = (const int*)d_in[3];
    const int*   dst  = (const int*)d_in[4];
    const float* wjs[4] = { (const float*)d_in[5], (const float*)d_in[6],
                            (const float*)d_in[7], (const float*)d_in[8] }; // (l,k)=00,01,10,11
    const float* wq   = (const float*)d_in[9];
    const int WK_BASE = 10, WV_BASE = 50, SI_BASE = 90;

    // ---- workspace carve-out ----
    char* p = (char*)d_ws;
    size_t off = 0;
    auto allocF = [&](size_t n) { float* r = (float*)(p + off);       off += ((n * 4 + 255) / 256) * 256; return r; };
    auto allocH = [&](size_t n) { _Float16* r = (_Float16*)(p + off); off += ((n * 2 + 255) / 256) * 256; return r; };

    // zero-initialized region (accumulators) first
    float* ke     = allocF((size_t)kE * 32);   // [E,8,4]
    float* v0     = allocF((size_t)kE * 8);    // [E,8,1]
    float* v1     = allocF((size_t)kE * 24);   // [E,8,3]
    float* denom  = allocF(kN);
    float* dstats = allocF(8);
    float* statsE = allocF(8 * 64);            // 8 edge MLPs x [sum32|sumsq32]
    float* statsS = allocF(4 * 256);           // 4 si BN layers x [sum128|sumsq128]
    size_t zero_bytes = off;

    float* scE    = allocF(8 * 64);            // per edge MLP [scale32|shift32]
    float* scS    = allocF(4 * 256);           // per si layer [scale128|shift128]
    _Float16* h1    = allocH((size_t)kE * 32);
    float*    h2pre = allocF((size_t)kE * 32);
    float*    Rbuf  = allocF((size_t)kE * 192);
    float*    qbuf  = allocF((size_t)kN * 32);
    float*    exb   = allocF(kE);
    _Float16* innerb = allocH((size_t)kN * 64);
    float*    sipre  = allocF((size_t)kN * 128);
    float*    sipre2 = allocF((size_t)kN * 128);
    float*    siw    = allocF((size_t)kN * 64);

    static const int mL[8]    = {0,0,1,1, 0,1,0,1};
    static const int mK[8]    = {0,1,0,1, 0,0,1,1};
    static const int mDout[8] = {64,64,64,192, 64,64,64,192};

    _Float16* w2hE[8]; _Float16* w3hE[8];
    for (int m = 0; m < 8; ++m) { w2hE[m] = allocH(32 * 32); w3hE[m] = allocH((size_t)mDout[m] * 32); }
    _Float16* siW1h[2]; _Float16* siW2h[2]; _Float16* siW3h[2];
    for (int l = 0; l < 2; ++l) { siW1h[l] = allocH(128 * 64); siW2h[l] = allocH(128 * 128); siW3h[l] = allocH(64 * 128); }

    float* out0 = (float*)d_out;               // [N,8,1]
    float* out1 = out0 + (size_t)kN * 8;       // [N,8,3]

    // ---- init ----
    hipMemsetAsync(d_out, 0, (size_t)out_size * sizeof(float), stream);
    hipMemsetAsync(d_ws, 0, zero_bytes, stream);

    // ---- weight f32->f16 (with K padding) ----
    for (int m = 0; m < 8; ++m) {
        int base = (m < 4) ? (WK_BASE + m * 10) : (WV_BASE + (m - 4) * 10);
        const float* W2 = (const float*)d_in[base + 1];
        const float* W3 = (const float*)d_in[base + 2];
        cvt_pad_f16<<<cdiv(32 * 32, 256), 256, 0, stream>>>(W2, 30, 10, w2hE[m], 32, 32 * 32);
        int t3 = mDout[m] * 32;
        cvt_pad_f16<<<cdiv(t3, 256), 256, 0, stream>>>(W3, mDout[m], 30, w3hE[m], 32, t3);
    }
    for (int l = 0; l < 2; ++l) {
        int base = SI_BASE + l * 10;
        cvt_pad_f16<<<cdiv(128 * 64, 256), 256, 0, stream>>>((const float*)d_in[base + 0], 128, 64,  siW1h[l], 64,  128 * 64);
        cvt_pad_f16<<<cdiv(128 * 128, 256), 256, 0, stream>>>((const float*)d_in[base + 1], 128, 128, siW2h[l], 128, 128 * 128);
        cvt_pad_f16<<<cdiv(64 * 128, 256), 256, 0, stream>>>((const float*)d_in[base + 2], 64,  128, siW3h[l], 128, 64 * 128);
    }

    // ---- scalar BN stats of dist ----
    reduce_dist_stats<<<256, 256, 0, stream>>>(dist, dstats);

    const int gE  = cdiv(kE, 256);
    const int gEW = kE / 128;   // 1250 (E % 128 == 0)
    const int gNW = kN / 128;   // 125  (N % 128 == 0)
    const int gN  = cdiv(kN, 256);

    // ---- 8 edge radial MLPs + CG contraction ----
    for (int m = 0; m < 8; ++m) {
        int base = (m < 4) ? (WK_BASE + m * 10) : (WV_BASE + (m - 4) * 10);
        const float* W1  = (const float*)d_in[base + 0];
        const float* b3  = (const float*)d_in[base + 5];
        const float* be1 = (const float*)d_in[base + 6];
        const float* be2 = (const float*)d_in[base + 7];
        const float* g1  = (const float*)d_in[base + 8];
        const float* g2  = (const float*)d_in[base + 9];
        float* st = statsE + m * 64;
        float* sc = scE + m * 64;

        edge_layer1<<<gE, 256, 0, stream>>>(dist, W1, g1, be1, dstats, h1);
        // layer2 GEMM with fused BN-stats epilogue
        wmma_gemm_xwT<32, 32, false, true><<<gEW, 256, 0, stream>>>(h1, w2hE[m], nullptr, h2pre, st, kE);
        bn_finalize<<<1, 32, 0, stream>>>(st, g2, be2, 32, 30, 1.f / kE, sc, sc + 32);
        // layer3 GEMM with BN+ReLU+cvt fused into the A-fragment load
        if (mDout[m] == 192)
            wmma_gemm_bnA<32, 192, true, false><<<gEW, 256, 0, stream>>>(h2pre, sc, sc + 32, w3hE[m], b3, Rbuf, nullptr, kE);
        else
            wmma_gemm_bnA<32, 64, true, false><<<gEW, 256, 0, stream>>>(h2pre, sc, sc + 32, w3hE[m], b3, Rbuf, nullptr, kE);

        const float* wj = wjs[mL[m] * 2 + mK[m]];
        switch (m) {
        case 0: edge_wlayer<0,0><<<gE,256,0,stream>>>(Rbuf, wj, f0, dst, ke, 4, 0); break;
        case 1: edge_wlayer<0,1><<<gE,256,0,stream>>>(Rbuf, wj, f1, dst, ke, 4, 0); break;
        case 2: edge_wlayer<1,0><<<gE,256,0,stream>>>(Rbuf, wj, f0, dst, ke, 4, 1); break;
        case 3: edge_wlayer<1,1><<<gE,256,0,stream>>>(Rbuf, wj, f1, dst, ke, 4, 1); break;
        case 4: edge_wlayer<0,0><<<gE,256,0,stream>>>(Rbuf, wj, f0, src, v0, 1, 0); break;
        case 5: edge_wlayer<1,0><<<gE,256,0,stream>>>(Rbuf, wj, f0, src, v1, 3, 0); break;
        case 6: edge_wlayer<0,1><<<gE,256,0,stream>>>(Rbuf, wj, f1, src, v0, 1, 0); break;
        case 7: edge_wlayer<1,1><<<gE,256,0,stream>>>(Rbuf, wj, f1, src, v1, 3, 0); break;
        }
    }

    // ---- attention + aggregation ----
    node_q<<<gN, 256, 0, stream>>>(f0, f1, wq, qbuf);
    edge_attn_ex<<<gE, 256, 0, stream>>>(qbuf, ke, dst, exb, denom);
    edge_agg<<<gE, 256, 0, stream>>>(exb, denom, dst, v0, v1, out0, out1);

    // ---- self-interaction MLPs ----
    for (int l = 0; l < 2; ++l) {
        int base = SI_BASE + l * 10;
        const float* b3  = (const float*)d_in[base + 5];
        const float* be1 = (const float*)d_in[base + 6];
        const float* be2 = (const float*)d_in[base + 7];
        const float* g1  = (const float*)d_in[base + 8];
        const float* g2  = (const float*)d_in[base + 9];
        float* st1 = statsS + (l * 2 + 0) * 256;
        float* st2 = statsS + (l * 2 + 1) * 256;
        float* sc1 = scS + (l * 2 + 0) * 256;
        float* sc2 = scS + (l * 2 + 1) * 256;

        if (l == 0) node_inner<1><<<gN, 256, 0, stream>>>(f0, innerb);
        else        node_inner<3><<<gN, 256, 0, stream>>>(f1, innerb);

        wmma_gemm_xwT<64, 128, false, true><<<gNW, 256, 0, stream>>>(innerb, siW1h[l], nullptr, sipre, st1, kN);
        bn_finalize<<<1, 128, 0, stream>>>(st1, g1, be1, 128, 128, 1.f / kN, sc1, sc1 + 128);

        wmma_gemm_bnA<128, 128, false, true><<<gNW, 256, 0, stream>>>(sipre, sc1, sc1 + 128, siW2h[l], nullptr, sipre2, st2, kN);
        bn_finalize<<<1, 128, 0, stream>>>(st2, g2, be2, 128, 128, 1.f / kN, sc2, sc2 + 128);

        wmma_gemm_bnA<128, 64, true, false><<<gNW, 256, 0, stream>>>(sipre2, sc2, sc2 + 128, siW3h[l], b3, siw, nullptr, kN);

        if (l == 0) node_final<1><<<gN, 256, 0, stream>>>(siw, f0, out0);
        else        node_final<3><<<gN, 256, 0, stream>>>(siw, f1, out1);
    }
}
